// ImageEnhancementRBFNet_85409719648965
// MI455X (gfx1250) — matrix operand order repaired and checked by reading, not compiled, not run on hardware
//
#include <hip/hip_runtime.h>
#include <hip/hip_bf16.h>

typedef __attribute__((ext_vector_type(16))) _Float16 v16h;
typedef __attribute__((ext_vector_type(8)))  float    v8f;

namespace {
constexpr int RN = 256;            // rbf neurons
constexpr int CN = 3;
constexpr int HN = 178, WN = 218;
constexpr int PN = HN * WN;        // 38804 pixels
constexpr int BN = 8;              // batch
constexpr int DN = CN * PN;        // 116412 fc outputs
constexpr int RB = 64;             // r's per block in act kernel (4 r-slabs)
}

// ---------------------------------------------------------------------------
// Zero the act[8][256] accumulator in workspace (d_ws is poisoned, not zeroed).
// ---------------------------------------------------------------------------
__global__ void zero_act_kernel(float* __restrict__ act) {
  int i = blockIdx.x * blockDim.x + threadIdx.x;
  if (i < BN * RN) act[i] = 0.0f;
}

// ---------------------------------------------------------------------------
// Phase 1: act[b][r] = sum_p exp(-||x[b,:,p]-c[r,:,p]||^2 * scale_r)
// One thread per pixel; 24 x-values held in registers; centers streamed once.
// Wave32 butterfly reduction -> LDS atomic per wave -> global atomic per block.
// ---------------------------------------------------------------------------
__global__ __launch_bounds__(256) void rbf_act_kernel(
    const float* __restrict__ x, const float* __restrict__ centers,
    const float* __restrict__ log_sigmas, float* __restrict__ act)
{
  __shared__ float act_lds[BN * RB];
  const int tid  = threadIdx.x;
  const int lane = tid & 31;
  const int p    = blockIdx.x * 256 + tid;
  const int r0   = blockIdx.y * RB;

  for (int i = tid; i < BN * RB; i += 256) act_lds[i] = 0.0f;
  __syncthreads();

  const bool valid = (p < PN);
  float xv[BN][CN];
  #pragma unroll
  for (int b = 0; b < BN; ++b)
    #pragma unroll
    for (int c = 0; c < CN; ++c)
      xv[b][c] = valid ? x[(b * CN + c) * PN + p] : 0.0f;

  for (int rr = 0; rr < RB; ++rr) {
    const int r = r0 + rr;
    // scale_r = -1 / (2 * exp(2*log_sigma))  (uniform -> scalar path)
    const float scale = -0.5f * __expf(-2.0f * log_sigmas[r]);
    float e[BN];
    if (valid) {
      const float c0 = centers[(size_t)(r * CN + 0) * PN + p];
      const float c1 = centers[(size_t)(r * CN + 1) * PN + p];
      const float c2 = centers[(size_t)(r * CN + 2) * PN + p];
      if (rr + 1 < RB)  // global_prefetch_b8 for next neuron's row
        __builtin_prefetch(&centers[(size_t)((r + 1) * CN) * PN + p], 0, 0);
      #pragma unroll
      for (int b = 0; b < BN; ++b) {
        const float d0 = xv[b][0] - c0;
        const float d1 = xv[b][1] - c1;
        const float d2 = xv[b][2] - c2;
        const float dd = d0 * d0 + d1 * d1 + d2 * d2;
        e[b] = __expf(dd * scale);
      }
    } else {
      #pragma unroll
      for (int b = 0; b < BN; ++b) e[b] = 0.0f;
    }
    // wave32 reduction over pixels, then one LDS atomic per wave
    #pragma unroll
    for (int b = 0; b < BN; ++b) {
      float v = e[b];
      #pragma unroll
      for (int m = 16; m >= 1; m >>= 1) v += __shfl_xor(v, m, 32);
      if (lane == 0) atomicAdd(&act_lds[b * RB + rr], v);
    }
  }
  __syncthreads();
  for (int i = tid; i < BN * RB; i += 256) {
    const int b = i / RB, rr = i % RB;
    atomicAdd(&act[b * RN + r0 + rr], act_lds[i]);
  }
}

// ---------------------------------------------------------------------------
// Phase 2: out[b][d] = sigmoid(sum_r act[b][r] * fc_w[d][r] + fc_b[d])
// WMMA f32<-f16 16x16x32: M=batch (padded 8->16), N=16 fc outputs per wave,
// K=256 in 8 steps. A-fragments (act) staged once per block in LDS.
// ---------------------------------------------------------------------------
__global__ __launch_bounds__(256) void fc_sigmoid_wmma_kernel(
    const float* __restrict__ act, const float* __restrict__ fc_w,
    const float* __restrict__ fc_b, float* __restrict__ out)
{
  __shared__ v16h a_lds[8 * 32];   // 8 k-steps x 32 lanes, 8 KB
  const int tid  = threadIdx.x;
  const int lane = tid & 31;
  const int wid  = tid >> 5;

  // Cooperative A-fragment build: contiguous k order used here for A 16x32 (same as B)
  //   M = lane%16;  K = (lane>=16 ? 16:0) + idx
  {
    const int kiter = wid;              // one k-step per wave
    const int m     = lane & 15;
    const int khiA  = (lane >> 4) * 16;
    v16h a;
    #pragma unroll
    for (int idx = 0; idx < 16; ++idx) {
      const int K = khiA + idx;
      const float av = (m < BN) ? act[m * RN + kiter * 32 + K] : 0.0f;
      a[idx] = (_Float16)av;
    }
    a_lds[kiter * 32 + lane] = a;
  }
  __syncthreads();

  const int tile = blockIdx.x * 8 + wid;   // one 16-wide N tile per wave
  const int d0   = tile * 16;
  if (d0 >= DN) return;                    // wave-uniform exit (after barrier)

  // B 32x16 f16, contiguous k order as in A: N = lane%16; lanes 0-15 hold K=0..15, lanes 16-31 K=16..31
  const int n    = d0 + (lane & 15);
  const int khiB = (lane >> 4) * 16;

  v8f acc = {};
  #pragma unroll
  for (int kk = 0; kk < 8; ++kk) {
    const int k0 = kk * 32;
    v16h a = a_lds[kk * 32 + lane];
    v16h bf;
    if (n < DN) {
      const float4* rowp =
          reinterpret_cast<const float4*>(fc_w + (size_t)n * RN + k0 + khiB);
      const float4 q0 = rowp[0];
      const float4 q1 = rowp[1];
      const float4 q2 = rowp[2];
      const float4 q3 = rowp[3];
      bf[0]  = (_Float16)q0.x; bf[1]  = (_Float16)q0.y;
      bf[2]  = (_Float16)q0.z; bf[3]  = (_Float16)q0.w;
      bf[4]  = (_Float16)q1.x; bf[5]  = (_Float16)q1.y;
      bf[6]  = (_Float16)q1.z; bf[7]  = (_Float16)q1.w;
      bf[8]  = (_Float16)q2.x; bf[9]  = (_Float16)q2.y;
      bf[10] = (_Float16)q2.z; bf[11] = (_Float16)q2.w;
      bf[12] = (_Float16)q3.x; bf[13] = (_Float16)q3.y;
      bf[14] = (_Float16)q3.z; bf[15] = (_Float16)q3.w;
    } else {
      #pragma unroll
      for (int idx = 0; idx < 16; ++idx) bf[idx] = (_Float16)0.0f;
    }
    // (neg_a, A, neg_b, B, c_mod, C, reuse_a, reuse_b)
    acc = __builtin_amdgcn_wmma_f32_16x16x32_f16(
        false, a, false, bf, (short)0, acc, false, false);
  }

  // C/D 16x16 f32 layout: VGPR i -> (M=i, N=lane) on lanes 0-15; M=8..15 are padding
  if (lane < 16) {
    const int d = d0 + lane;
    if (d < DN) {
      const float bias = fc_b[d];
      #pragma unroll
      for (int b = 0; b < BN; ++b) {
        const float v = acc[b] + bias;
        out[(size_t)b * DN + d] = 1.0f / (1.0f + __expf(-v));
      }
    }
  }
}

// ---------------------------------------------------------------------------
extern "C" void kernel_launch(void* const* d_in, const int* in_sizes, int n_in,
                              void* d_out, int out_size, void* d_ws, size_t ws_size,
                              hipStream_t stream) {
  const float* x       = (const float*)d_in[0];  // [8,3,178,218]
  const float* centers = (const float*)d_in[1];  // [256,3,178,218]
  const float* lsig    = (const float*)d_in[2];  // [256]
  const float* fc_w    = (const float*)d_in[3];  // [116412,256]
  const float* fc_b    = (const float*)d_in[4];  // [116412]
  float* out = (float*)d_out;                    // [8,116412] f32
  float* act = (float*)d_ws;                     // 8*256 floats = 8 KB scratch

  zero_act_kernel<<<(BN * RN + 255) / 256, 256, 0, stream>>>(act);

  dim3 gA((PN + 255) / 256, RN / RB);            // 152 x 4 blocks
  rbf_act_kernel<<<gA, 256, 0, stream>>>(x, centers, lsig, act);

  const int tiles = (DN + 15) / 16;              // 7276 column tiles
  fc_sigmoid_wmma_kernel<<<(tiles + 7) / 8, 256, 0, stream>>>(act, fc_w, fc_b, out);
}
